// WaveGlow_16716012716216
// MI455X (gfx1250) — matrix operand order repaired and checked by reading, not compiled, not run on hardware
//
#include <hip/hip_runtime.h>
#include <hip/hip_bf16.h>
#include <math.h>

// ---------------------------------------------------------------------------
// WaveGlow inference for MI455X (gfx1250, wave32, WMMA).
//
// Input flattening assumption (jax pytree order: top-level dict insertion
// order from setup_inputs, nested dicts sorted by key, lists by index):
//   0: spect (1,80,F)   1: z0 (1,4,T)   2: z8 (1,2,T)   3: z4 (1,2,T)
//   4..15 : params['inv'][0..11]
//   16: up_b (80,)      17: up_w (80,80,1024)
//   18+12k+j : params['wn'][k] leaves, keys sorted:
//     0 cond_b  1 cond_w  2 end_b  3 end_w  4 in_b  5 in_w
//     6 res_b   7 res_w   8 skip_b 9 skip_w 10 start_b 11 start_w
// ---------------------------------------------------------------------------

typedef __attribute__((ext_vector_type(16))) _Float16 v16h;
typedef __attribute__((ext_vector_type(8)))  _Float16 v8h;
typedef __attribute__((ext_vector_type(2)))  _Float16 v2h;
typedef __attribute__((ext_vector_type(8)))  float    v8f;
typedef __attribute__((ext_vector_type(2)))  float    v2f;

#define PAD 128   // max dilation (2^7); zero-padded rows around x_f16

// ---- WMMA fragment loaders (contiguous K order k = 16*half + e, same for A and B) ----
// A (16x32, MxK): lanes 0-15: M=lane, elems = K{k0..k0+15};
//                 lanes16-31: M=lane-16, elems = K{k0+16..k0+31}.
__device__ __forceinline__ v16h load_a_frag(const _Float16* __restrict__ W,
                                            int ldw, int m0, int k0, int lane) {
    const int half = lane >> 4;
    const int r    = lane & 15;
    const _Float16* p = W + (size_t)(m0 + r) * ldw + k0 + half * 16;
    v8h lo = *(const v8h*)p;         // K = k0+half*16 .. +7
    v8h hi = *(const v8h*)(p + 8);   // K = k0+half*16+8 .. +7
    v16h a;
#pragma unroll
    for (int e = 0; e < 8; ++e) { a[e] = lo[e]; a[e + 8] = hi[e]; }
    return a;
}
// B (32x16, KxN): lanes 0-15: N=lane, elems = K{k0..k0+15};
//                 lanes16-31: N=lane-16, elems = K{k0+16..k0+31}.
// With time-major activations ([T][C], C contiguous) this is one 32B load.

__device__ __forceinline__ v8f wmma_f16(v16h a, v16h b, v8f c) {
    return __builtin_amdgcn_wmma_f32_16x16x32_f16(false, a, false, b,
                                                  (short)0, c, false, false);
}

__device__ __forceinline__ float sigm(float x) {
    return 1.0f / (1.0f + __expf(-x));
}

// ---------------------------------------------------------------------------
// Fused dilated-in-conv + cond-conv GEMM with gated-tanh epilogue.
// K = [tap0 | tap1 | tap2 | cond] = 256*3 + 640 = 1408.
// xin points into a zero-padded [PAD + T + PAD][256] buffer, so tap loads
// need no bounds checks (branch-free inner loop).
// Each wave: one 16-row M tile (both tanh & sigmoid halves) x two 16-col
// N tiles -> 4 accumulators, A fragments reused across both N tiles.
// ---------------------------------------------------------------------------
__global__ void __launch_bounds__(256) gemm_acts_kernel(
    const _Float16* __restrict__ Wpk,   // [512][1408] f16, packed per layer
    const _Float16* __restrict__ xin,   // [T][256] f16 (padded base + PAD*256)
    const _Float16* __restrict__ sp,    // [T][640] f16
    const float*    __restrict__ in_b,  // [512]
    const float*    __restrict__ cond_b,// [512]
    _Float16*       __restrict__ zout,  // [T][256] f16
    int T, int dil)
{
    const int lane = threadIdx.x & 31;
    const int wid  = (int)((blockIdx.x * blockDim.x + threadIdx.x) >> 5);
    const int mt = wid & 15;         // 16 M tiles -> rows 0..255 (+256 pair)
    const int np = wid >> 4;         // N-pair tile
    if (np >= (T >> 5)) return;
    const int m0   = mt << 4;
    const int half = lane >> 4;
    const int ln   = lane & 15;
    const int t0   = (np << 5) + ln;     // column of first N tile
    const int t1   = t0 + 16;            // column of second N tile

    v8f c00 = {}, c01 = {};   // tanh rows    x {t0,t1}
    v8f c10 = {}, c11 = {};   // sigmoid rows x {t0,t1}

    // --- three dilated taps of x (8 K-steps each, branch-free) ---
#pragma unroll
    for (int seg = 0; seg < 3; ++seg) {
        const _Float16* b0 =
            xin + (size_t)(t0 + (seg - 1) * dil) * 256 + half * 16;
        for (int ks = 0; ks < 8; ++ks) {
            const int k0 = seg * 256 + ks * 32;
            v16h bA = *(const v16h*)(b0 + ks * 32);
            v16h bB = *(const v16h*)(b0 + 16 * 256 + ks * 32);
            v16h at = load_a_frag(Wpk, 1408, m0,       k0, lane);
            v16h as = load_a_frag(Wpk, 1408, m0 + 256, k0, lane);
            c00 = wmma_f16(at, bA, c00);
            c01 = wmma_f16(at, bB, c01);
            c10 = wmma_f16(as, bA, c10);
            c11 = wmma_f16(as, bB, c11);
        }
    }
    // --- conditioning part (20 K-steps) ---
    {
        const _Float16* s0 = sp + (size_t)t0 * 640 + half * 16;
        for (int ks = 0; ks < 20; ++ks) {
            const int k0 = 768 + ks * 32;
            v16h bA = *(const v16h*)(s0 + ks * 32);
            v16h bB = *(const v16h*)(s0 + 16 * 640 + ks * 32);
            v16h at = load_a_frag(Wpk, 1408, m0,       k0, lane);
            v16h as = load_a_frag(Wpk, 1408, m0 + 256, k0, lane);
            c00 = wmma_f16(at, bA, c00);
            c01 = wmma_f16(at, bB, c01);
            c10 = wmma_f16(as, bA, c10);
            c11 = wmma_f16(as, bB, c11);
        }
    }

    // C layout: elem e -> row m0 + half*8 + e, col = lane&15.
    // Adjacent e => adjacent channels -> pack pairs into 4-byte stores.
#pragma unroll
    for (int e = 0; e < 8; e += 2) {
        const int m = m0 + half * 8 + e;
        const float bt0 = in_b[m]       + cond_b[m];
        const float bt1 = in_b[m + 1]   + cond_b[m + 1];
        const float bs0 = in_b[m + 256] + cond_b[m + 256];
        const float bs1 = in_b[m + 257] + cond_b[m + 257];
        v2h q0, q1;
        q0[0] = (_Float16)(tanhf(c00[e]     + bt0) * sigm(c10[e]     + bs0));
        q0[1] = (_Float16)(tanhf(c00[e + 1] + bt1) * sigm(c10[e + 1] + bs1));
        q1[0] = (_Float16)(tanhf(c01[e]     + bt0) * sigm(c11[e]     + bs0));
        q1[1] = (_Float16)(tanhf(c01[e + 1] + bt1) * sigm(c11[e + 1] + bs1));
        *(v2h*)(zout + (size_t)t0 * 256 + m) = q0;
        *(v2h*)(zout + (size_t)t1 * 256 + m) = q1;
    }
}

// ---------------------------------------------------------------------------
// Fused res+skip 1x1 convs. Wrs rows 0..255 = res, 256..511 = skip.
// x += res(z)+res_b (if has_res, fp32 master + f16 copy); skip += skip(z)+b.
// ---------------------------------------------------------------------------
__global__ void __launch_bounds__(256) gemm_res_skip_kernel(
    const _Float16* __restrict__ Wrs,    // [512][256] f16
    const _Float16* __restrict__ z,      // [T][256] f16
    const float*    __restrict__ res_b,  // [256]
    const float*    __restrict__ skip_b, // [256]
    float*          __restrict__ xf32,   // [T][256]
    _Float16*       __restrict__ xf16,   // [T][256] (padded base + PAD*256)
    float*          __restrict__ skipbuf,// [T][256]
    int T, int has_res)
{
    const int lane = threadIdx.x & 31;
    const int wid  = (int)((blockIdx.x * blockDim.x + threadIdx.x) >> 5);
    const int mt = wid & 15, np = wid >> 4;
    if (np >= (T >> 5)) return;
    const int m0   = mt << 4;
    const int half = lane >> 4;
    const int ln   = lane & 15;
    const int t0   = (np << 5) + ln;
    const int t1   = t0 + 16;

    v8f cr0 = {}, cr1 = {}, cs0 = {}, cs1 = {};
    const _Float16* zb = z + (size_t)t0 * 256 + half * 16;
#pragma unroll
    for (int ks = 0; ks < 8; ++ks) {
        const int k0 = ks << 5;
        v16h bA = *(const v16h*)(zb + ks * 32);
        v16h bB = *(const v16h*)(zb + 16 * 256 + ks * 32);
        v16h as = load_a_frag(Wrs, 256, m0 + 256, k0, lane);
        cs0 = wmma_f16(as, bA, cs0);
        cs1 = wmma_f16(as, bB, cs1);
        if (has_res) {
            v16h ar = load_a_frag(Wrs, 256, m0, k0, lane);
            cr0 = wmma_f16(ar, bA, cr0);
            cr1 = wmma_f16(ar, bB, cr1);
        }
    }
#pragma unroll
    for (int e = 0; e < 8; e += 2) {
        const int m = m0 + half * 8 + e;
        const size_t o0 = (size_t)t0 * 256 + m;
        const size_t o1 = (size_t)t1 * 256 + m;
        const float sb0 = skip_b[m], sb1 = skip_b[m + 1];
        v2f s0 = *(const v2f*)(skipbuf + o0);
        v2f s1 = *(const v2f*)(skipbuf + o1);
        s0[0] += cs0[e] + sb0;  s0[1] += cs0[e + 1] + sb1;
        s1[0] += cs1[e] + sb0;  s1[1] += cs1[e + 1] + sb1;
        *(v2f*)(skipbuf + o0) = s0;
        *(v2f*)(skipbuf + o1) = s1;
        if (has_res) {
            const float rb0 = res_b[m], rb1 = res_b[m + 1];
            v2f x0 = *(const v2f*)(xf32 + o0);
            v2f x1 = *(const v2f*)(xf32 + o1);
            x0[0] += cr0[e] + rb0;  x0[1] += cr0[e + 1] + rb1;
            x1[0] += cr1[e] + rb0;  x1[1] += cr1[e + 1] + rb1;
            *(v2f*)(xf32 + o0) = x0;
            *(v2f*)(xf32 + o1) = x1;
            v2h h0, h1;
            h0[0] = (_Float16)x0[0]; h0[1] = (_Float16)x0[1];
            h1[0] = (_Float16)x1[0]; h1[1] = (_Float16)x1[1];
            *(v2h*)(xf16 + o0) = h0;
            *(v2h*)(xf16 + o1) = h1;
        }
    }
}

// ---------------------------------------------------------------------------
// Weight packing (fp32 -> f16, tap-major K ordering for contiguous A rows)
// ---------------------------------------------------------------------------
__global__ void pack_acts_w(const float* __restrict__ in_w,   // [8][512][256][3]
                            const float* __restrict__ cond_w, // [8][512][640]
                            int layer, _Float16* __restrict__ Wpk) // [512][1408]
{
    int idx = blockIdx.x * blockDim.x + threadIdx.x;
    if (idx >= 512 * 1408) return;
    int c = idx / 1408, k = idx - c * 1408;
    float v;
    if (k < 768) {
        int tap = k >> 8, cin = k & 255;
        v = in_w[(((size_t)layer * 512 + c) * 256 + cin) * 3 + tap];
    } else {
        v = cond_w[((size_t)layer * 512 + c) * 640 + (k - 768)];
    }
    Wpk[idx] = (_Float16)v;
}

__global__ void pack_rs_w(const float* __restrict__ res_w,   // [7][256][256]
                          const float* __restrict__ skip_w,  // [8][256][256]
                          int layer, int has_res, _Float16* __restrict__ Wrs)
{
    int idx = blockIdx.x * blockDim.x + threadIdx.x;
    if (idx >= 512 * 256) return;
    int r = idx >> 8, cin = idx & 255;
    float v;
    if (r < 256)
        v = has_res ? res_w[((size_t)layer * 256 + r) * 256 + cin] : 0.0f;
    else
        v = skip_w[((size_t)layer * 256 + (r - 256)) * 256 + cin];
    Wrs[idx] = (_Float16)v;
}

// ---------------------------------------------------------------------------
// Upsample transposed conv -> conditioning sp[t][c*8+g] (f16)
// up[c,l] = up_b[c] + sum_{j=0..3, t=floor(l/256)-j >= 0} sum_cin
//           spect[cin,t] * up_w[cin,c, l - 256 t]
// ---------------------------------------------------------------------------
__global__ void upsample_kernel(const float* __restrict__ spect,
                                const float* __restrict__ up_w,
                                const float* __restrict__ up_b,
                                _Float16* __restrict__ sp,
                                int frames, int L)
{
    int idx = blockIdx.x * blockDim.x + threadIdx.x;
    if (idx >= 80 * L) return;
    int c = idx / L, l = idx - c * L;
    float acc = up_b[c];
    const int tbase = l >> 8, hbase = l & 255;
    for (int j = 0; j < 4; ++j) {
        int tt = tbase - j;
        if (tt < 0) break;
        int h = hbase + (j << 8);
        const float* wp  = up_w + (size_t)c * 1024 + h;   // + cin*80*1024
        const float* spp = spect + tt;                    // + cin*frames
        float s = 0.f;
        for (int cin = 0; cin < 80; ++cin)
            s += spp[(size_t)cin * frames] * wp[(size_t)cin * 80 * 1024];
        acc += s;
    }
    int t = l >> 3, g = l & 7;
    sp[(size_t)t * 640 + c * 8 + g] = (_Float16)acc;
}

// ---- zero the f16 padding halo (once per call) ----------------------------
__global__ void zero_pad_kernel(_Float16* __restrict__ xf16_base, int T)
{
    int i = blockIdx.x * blockDim.x + threadIdx.x;
    if (i >= 2 * PAD * 256) return;
    int t = i >> 8, c = i & 255;                      // t in [0, 2*PAD)
    int row = (t < PAD) ? t : (PAD + T + (t - PAD));  // front / back halo
    xf16_base[(size_t)row * 256 + c] = (_Float16)0.0f;
}

// ---- start conv (nh->256) + zero skip accumulator -------------------------
__global__ void start_conv_kernel(const float* __restrict__ audio, int a0_off,
                                  int nh, int T,
                                  const float* __restrict__ start_w, // [256][nh]
                                  const float* __restrict__ start_b,
                                  float* __restrict__ xf32,
                                  _Float16* __restrict__ xf16,  // +PAD*256 base
                                  float* __restrict__ skipbuf)
{
    int idx = blockIdx.x * blockDim.x + threadIdx.x;
    if (idx >= T * 256) return;
    int t = idx >> 8, c = idx & 255;
    float acc = start_b[c];
    for (int j = 0; j < nh; ++j)
        acc += audio[(size_t)(a0_off + j) * T + t] * start_w[c * nh + j];
    xf32[idx] = acc;
    xf16[idx] = (_Float16)acc;
    skipbuf[idx] = 0.0f;
}

// ---- end conv (256 -> 2nh) ------------------------------------------------
__global__ void end_conv_kernel(const float* __restrict__ skipbuf,
                                const float* __restrict__ end_w, // [2nh][256]
                                const float* __restrict__ end_b,
                                float* __restrict__ endout,      // [T][2nh]
                                int n, int T)
{
    int idx = blockIdx.x * blockDim.x + threadIdx.x;
    if (idx >= T * n) return;
    int t = idx / n, j = idx - t * n;
    float acc = end_b[j];
    const float* w = end_w + (size_t)j * 256;
    const float* s = skipbuf + (size_t)t * 256;
    for (int c = 0; c < 256; ++c) acc += w[c] * s[c];
    endout[idx] = acc;
}

// ---- affine coupling + invertible 1x1 mix (+ optional z concat) -----------
__global__ void couple_mix_kernel(const float* __restrict__ endout,
                                  const float* __restrict__ ain,
                                  float* __restrict__ aout,
                                  const float* __restrict__ winv, // [8][8] padded
                                  int nh, int parity,
                                  const float* __restrict__ zcat, int T)
{
    int t = blockIdx.x * blockDim.x + threadIdx.x;
    if (t >= T) return;
    const int n = 2 * nh;
    float cat[8];
    const float* a0 = ain + (size_t)(parity ? nh : 0) * T;
    const float* a1 = ain + (size_t)(parity ? 0 : nh) * T;
    for (int j = 0; j < nh; ++j) {
        float b = endout[(size_t)t * n + j];
        float s = endout[(size_t)t * n + nh + j];
        float v = (a1[(size_t)j * T + t] - b) * __expf(-s);
        float u = a0[(size_t)j * T + t];
        if (parity == 0) { cat[j] = u; cat[nh + j] = v; }   // [a0, a1']
        else             { cat[j] = v; cat[nh + j] = u; }   // [a1', a0]
    }
    const int off = zcat ? 2 : 0;
    if (zcat) {
        aout[t]             = zcat[t];              // SIGMA == 1.0
        aout[(size_t)T + t] = zcat[(size_t)T + t];
    }
    for (int i = 0; i < n; ++i) {
        const float* wr = winv + i * 8;
        float acc = 0.f;
        for (int j = 0; j < n; ++j) acc += wr[j] * cat[j];
        aout[(size_t)(off + i) * T + t] = acc;
    }
}

// ---- tiny Gauss-Jordan inverse (one thread per flow) ----------------------
struct InvArgs { const float* m[12]; int n[12]; };

__global__ void invert_kernel(InvArgs args, float* __restrict__ winv)
{
    int f = threadIdx.x;
    if (f >= 12) return;
    const int n = args.n[f];
    float a[8][16];
    for (int i = 0; i < n; ++i)
        for (int j = 0; j < n; ++j) {
            a[i][j]     = args.m[f][i * n + j];
            a[i][n + j] = (i == j) ? 1.f : 0.f;
        }
    for (int col = 0; col < n; ++col) {
        int piv = col; float best = fabsf(a[col][col]);
        for (int r = col + 1; r < n; ++r) {
            float v = fabsf(a[r][col]);
            if (v > best) { best = v; piv = r; }
        }
        if (piv != col)
            for (int j = 0; j < 2 * n; ++j) {
                float tmp = a[col][j]; a[col][j] = a[piv][j]; a[piv][j] = tmp;
            }
        float inv = 1.0f / a[col][col];
        for (int j = 0; j < 2 * n; ++j) a[col][j] *= inv;
        for (int r = 0; r < n; ++r) if (r != col) {
            float fr = a[r][col];
            for (int j = 0; j < 2 * n; ++j) a[r][j] -= fr * a[col][j];
        }
    }
    float* out = winv + f * 64;
    for (int i = 0; i < 8; ++i)
        for (int j = 0; j < 8; ++j)
            out[i * 8 + j] = (i < n && j < n) ? a[i][n + j] : 0.f;
}

__global__ void init_audio_kernel(const float* __restrict__ z0,
                                  float* __restrict__ audio, int n)
{
    int i = blockIdx.x * blockDim.x + threadIdx.x;
    if (i < n) audio[i] = z0[i];   // SIGMA == 1.0
}

__global__ void final_out_kernel(const float* __restrict__ audio,
                                 float* __restrict__ out, int T)
{
    int i = blockIdx.x * blockDim.x + threadIdx.x;
    if (i >= 8 * T) return;
    int t = i >> 3, c = i & 7;
    out[i] = audio[(size_t)c * T + t];
}

// ---------------------------------------------------------------------------
extern "C" void kernel_launch(void* const* d_in, const int* in_sizes, int n_in,
                              void* d_out, int out_size, void* d_ws, size_t ws_size,
                              hipStream_t stream)
{
    (void)n_in; (void)out_size; (void)ws_size;
    const float* spect = (const float*)d_in[0];
    const float* z0    = (const float*)d_in[1];
    const float* z8    = (const float*)d_in[2];
    const float* z4    = (const float*)d_in[3];
    const float* up_b  = (const float*)d_in[16];
    const float* up_w  = (const float*)d_in[17];

    struct Flow {
        const float *cond_b, *cond_w, *end_b, *end_w, *in_b, *in_w,
                    *res_b, *res_w, *skip_b, *skip_w, *start_b, *start_w;
    } fl[12];
    for (int k = 0; k < 12; ++k) {
        const int b = 18 + k * 12;
        fl[k].cond_b  = (const float*)d_in[b + 0];
        fl[k].cond_w  = (const float*)d_in[b + 1];
        fl[k].end_b   = (const float*)d_in[b + 2];
        fl[k].end_w   = (const float*)d_in[b + 3];
        fl[k].in_b    = (const float*)d_in[b + 4];
        fl[k].in_w    = (const float*)d_in[b + 5];
        fl[k].res_b   = (const float*)d_in[b + 6];
        fl[k].res_w   = (const float*)d_in[b + 7];
        fl[k].skip_b  = (const float*)d_in[b + 8];
        fl[k].skip_w  = (const float*)d_in[b + 9];
        fl[k].start_b = (const float*)d_in[b + 10];
        fl[k].start_w = (const float*)d_in[b + 11];
    }

    const int frames = in_sizes[0] / 80;   // 100
    const int L = frames * 256;            // 25600
    const int T = L / 8;                   // 3200 (multiple of 32)

    // ---- workspace carve (aligned to 256B) ----
    size_t off = 0;
    auto carve = [&](size_t bytes) -> char* {
        char* p = (char*)d_ws + off;
        off = (off + bytes + 255) & ~(size_t)255;
        return p;
    };
    _Float16* sp16      = (_Float16*)carve((size_t)T * 640 * 2);
    float*    xf32      = (float*)   carve((size_t)T * 256 * 4);
    _Float16* xf16_base = (_Float16*)carve((size_t)(T + 2 * PAD) * 256 * 2);
    _Float16* z16       = (_Float16*)carve((size_t)T * 256 * 2);
    float*    skipbuf   = (float*)   carve((size_t)T * 256 * 4);
    float*    endout    = (float*)   carve((size_t)T * 8 * 4);
    float*    audA      = (float*)   carve((size_t)8 * T * 4);
    float*    audB      = (float*)   carve((size_t)8 * T * 4);
    float*    winv      = (float*)   carve((size_t)12 * 64 * 4);
    _Float16* Wpk       = (_Float16*)carve((size_t)512 * 1408 * 2);
    _Float16* Wrs       = (_Float16*)carve((size_t)512 * 256 * 2);
    _Float16* xf16      = xf16_base + (size_t)PAD * 256;  // interior view

    // ---- invert the 1x1 mixing matrices ----
    static const int NREM[12] = {8,8,8,8,6,6,6,6,4,4,4,4};
    InvArgs ia;
    for (int k = 0; k < 12; ++k) { ia.m[k] = (const float*)d_in[4 + k]; ia.n[k] = NREM[k]; }
    invert_kernel<<<1, 16, 0, stream>>>(ia, winv);

    // ---- zero padding halo, upsample conditioning, audio = z0 ----
    zero_pad_kernel<<<(2 * PAD * 256 + 255) / 256, 256, 0, stream>>>(xf16_base, T);
    {
        int total = 80 * L;
        upsample_kernel<<<(total + 255) / 256, 256, 0, stream>>>(
            spect, up_w, up_b, sp16, frames, L);
    }
    init_audio_kernel<<<(4 * T + 255) / 256, 256, 0, stream>>>(z0, audA, 4 * T);

    float* aCur = audA; float* aNext = audB;
    int cur_ch = 4;
    const int gemm_blocks = (16 * (T / 32)) / 8;   // 8 waves per 256-thread block

    for (int k = 11; k >= 0; --k) {
        const Flow& F = fl[k];
        const int nh = cur_ch / 2;
        const int n  = cur_ch;
        const int parity = k & 1;
        const int a0_off = parity ? nh : 0;

        start_conv_kernel<<<(T * 256 + 255) / 256, 256, 0, stream>>>(
            aCur, a0_off, nh, T, F.start_w, F.start_b, xf32, xf16, skipbuf);

        for (int i = 0; i < 8; ++i) {
            const int has_res = (i < 7) ? 1 : 0;
            pack_acts_w<<<(512 * 1408 + 255) / 256, 256, 0, stream>>>(
                F.in_w, F.cond_w, i, Wpk);
            pack_rs_w<<<(512 * 256 + 255) / 256, 256, 0, stream>>>(
                F.res_w, F.skip_w, i, has_res, Wrs);
            gemm_acts_kernel<<<gemm_blocks, 256, 0, stream>>>(
                Wpk, xf16, sp16, F.in_b + i * 512, F.cond_b + i * 512,
                z16, T, 1 << i);
            gemm_res_skip_kernel<<<gemm_blocks, 256, 0, stream>>>(
                Wrs, z16,
                has_res ? F.res_b + i * 256 : F.res_b,
                F.skip_b + i * 256,
                xf32, xf16, skipbuf, T, has_res);
        }

        end_conv_kernel<<<(T * n + 255) / 256, 256, 0, stream>>>(
            skipbuf, F.end_w, F.end_b, endout, n, T);

        const float* zc = (k == 8) ? z8 : (k == 4) ? z4 : nullptr;
        couple_mix_kernel<<<(T + 255) / 256, 256, 0, stream>>>(
            endout, aCur, aNext, winv + k * 64, nh, parity, zc, T);

        float* tmp = aCur; aCur = aNext; aNext = tmp;
        if (zc) cur_ch = n + 2;
    }

    final_out_kernel<<<(8 * T + 255) / 256, 256, 0, stream>>>(
        aCur, (float*)d_out, T);
}
